// MultiHeadAttention_2542620639285
// MI455X (gfx1250) — matrix operand order repaired and checked by reading, not compiled, not run on hardware
//
#include <hip/hip_runtime.h>
#include <hip/hip_bf16.h>

typedef _Float16 h8  __attribute__((ext_vector_type(8)));
typedef _Float16 h16 __attribute__((ext_vector_type(16)));
typedef float    f8  __attribute__((ext_vector_type(8)));

#define T_SEQ 2048
#define C_DIM 1024
#define H_NUM 16
#define D_HEAD 64
#define B_BATCH 4
#define M_ROWS (B_BATCH * T_SEQ)   // 8192
// scale folded with log2(e): softmax computed in exp2 domain (HW v_exp_f32 is exp2)
#define ATTN_SCALE_LOG2 (0.125f * 1.4426950408889634f)

// ---------------------------------------------------------------------------
// WMMA wrapper: D(f32 16x16) = A(f16 16x32) * B(f16 32x16) + C
// ---------------------------------------------------------------------------
__device__ __forceinline__ f8 wmma_f16(h16 a, h16 b, f8 c) {
  return __builtin_amdgcn_wmma_f32_16x16x32_f16(
      /*neg_a=*/false, a, /*neg_b=*/false, b,
      /*c_mod=*/(short)0, c, /*reuse_a=*/false, /*reuse_b=*/false);
}

// A-fragment (16x32 f16) from row-major f16, row stride ld.
// lane L holds row (L&15); halves [0..7]=K{kb..kb+7}, [8..15]=K{kb+16..kb+23},
// kb = k0 + 8*(L>=16).   (ISA 7.12.2)
__device__ __forceinline__ h16 load_a_h16(const _Float16* __restrict__ p,
                                          int ld, int row0, int k0) {
  int lane = threadIdx.x & 31;
  int r  = row0 + (lane & 15);
  int kb = k0 + ((lane & 16) >> 1);
  const _Float16* q = p + (size_t)r * ld + kb;
  h8 lo = *(const h8*)(q);
  h8 hi = *(const h8*)(q + 16);
  h16 f;
#pragma unroll
  for (int i = 0; i < 8; ++i) { f[i] = lo[i]; f[i + 8] = hi[i]; }
  return f;
}

// Same A-fragment sourced from f32 (converted in-register).
__device__ __forceinline__ h16 load_a_f32(const float* __restrict__ p,
                                          int ld, int row0, int k0) {
  int lane = threadIdx.x & 31;
  int r  = row0 + (lane & 15);
  int kb = k0 + ((lane & 16) >> 1);
  const float* q = p + (size_t)r * ld + kb;
  h16 f;
#pragma unroll
  for (int i = 0; i < 8; ++i) {
    f[i]     = (_Float16)q[i];
    f[i + 8] = (_Float16)q[16 + i];
  }
  return f;
}

// B-fragment (32x16): column n = col0 + (lane&15); halves [0..7]=K{kb..kb+7},
// [8..15]=K{kb+16..kb+23}, kb = k0 + 8*(L>=16) (same order as A).
__device__ __forceinline__ h16 load_b_h16(const _Float16* __restrict__ p,
                                          int ld, int col0, int k0) {
  int lane = threadIdx.x & 31;
  int n  = col0 + (lane & 15);
  int kb = k0 + ((lane & 16) >> 1);
  const _Float16* q = p + (size_t)n * ld + kb;
  h8 lo = *(const h8*)(q);
  h8 hi = *(const h8*)(q + 16);
  h16 f;
#pragma unroll
  for (int i = 0; i < 8; ++i) { f[i] = lo[i]; f[i + 8] = hi[i]; }
  return f;
}

// ---------------------------------------------------------------------------
// f32 -> f16 weight conversion
// ---------------------------------------------------------------------------
__global__ void cvt_f32_f16(const float* __restrict__ in,
                            _Float16* __restrict__ out, int n) {
  int i = blockIdx.x * blockDim.x + threadIdx.x;
  if (i < n) out[i] = (_Float16)in[i];
}

// ---------------------------------------------------------------------------
// GEMM: Out = A(M x K) * W(N x K)^T + bias   (NT, K-contiguous both sides)
// One wave computes a 32x64 tile: 2 A-fragments x 4 B-fragments -> 8 WMMAs
// per 32-step of K (16 b128 loads per 8 WMMAs).
// MODE 0: f16 out row-major (M x N)
// MODE 1: f16 out per-head transposed (b, h, d, t)    [for V]
// MODE 2: f32 out row-major (M x N)                   [final projection]
// ---------------------------------------------------------------------------
template <int MODE, typename AT>
__global__ __launch_bounds__(128)
void proj_gemm(const AT* __restrict__ A, const _Float16* __restrict__ W,
               const float* __restrict__ bias, void* __restrict__ Out) {
  const int K = C_DIM, N = C_DIM;
  int wid    = (blockIdx.x * blockDim.x + threadIdx.x) >> 5;  // global wave id
  int tile_n = wid & 15;    // N/64 = 16 column tiles
  int tile_m = wid >> 4;    // M/32 = 256 row tiles
  int m0 = tile_m * 32;
  int n0 = tile_n * 64;
  int lane = threadIdx.x & 31;

  f8 acc[2][4] = {};
  for (int k0 = 0; k0 < K; k0 += 32) {
    h16 a0, a1;
    if constexpr (sizeof(AT) == 4) {
      a0 = load_a_f32((const float*)A, K, m0,      k0);
      a1 = load_a_f32((const float*)A, K, m0 + 16, k0);
    } else {
      a0 = load_a_h16((const _Float16*)A, K, m0,      k0);
      a1 = load_a_h16((const _Float16*)A, K, m0 + 16, k0);
    }
#pragma unroll
    for (int j = 0; j < 4; ++j) {
      h16 b = load_b_h16(W, K, n0 + 16 * j, k0);
      acc[0][j] = wmma_f16(a0, b, acc[0][j]);
      acc[1][j] = wmma_f16(a1, b, acc[1][j]);
    }
  }

  int lg = (lane >> 4);      // half-group -> row offset 0/8
  int nn = (lane & 15);
#pragma unroll
  for (int r = 0; r < 2; ++r) {
    int m0r = m0 + 16 * r;
#pragma unroll
    for (int j = 0; j < 4; ++j) {
      int n = n0 + 16 * j + nn;
      float bv = bias[n];
      if constexpr (MODE == 0) {
        _Float16* O = (_Float16*)Out;
#pragma unroll
        for (int i = 0; i < 8; ++i) {
          int m = m0r + i + 8 * lg;
          O[(size_t)m * N + n] = (_Float16)(acc[r][j][i] + bv);
        }
      } else if constexpr (MODE == 1) {
        _Float16* O = (_Float16*)Out;
        int bb = m0r >> 11;                   // batch = m / T
        int hh = n >> 6;
        int dd = n & 63;
        size_t rowbase = (((size_t)(bb * H_NUM + hh) * D_HEAD) + dd) * T_SEQ;
        int tb = (m0r & (T_SEQ - 1)) + 8 * lg;
        h8 vals;
#pragma unroll
        for (int i = 0; i < 8; ++i) vals[i] = (_Float16)(acc[r][j][i] + bv);
        *(h8*)(O + rowbase + tb) = vals;      // 8 contiguous t -> one b128
      } else {
        float* O = (float*)Out;
#pragma unroll
        for (int i = 0; i < 8; ++i) {
          int m = m0r + i + 8 * lg;
          O[(size_t)m * N + n] = acc[r][j][i] + bv;
        }
      }
    }
  }
}

// ---------------------------------------------------------------------------
// Flash attention: one wave per (b, h, 16-query tile); streams keys in chunks
// of 64 with online softmax (exp2 domain). Q,K (B,T,C) f16; V transposed
// (b,h,d,t) f16. 16 WMMAs per 64-key chunk vs 64 cross-lane reduce ops.
// ---------------------------------------------------------------------------
__global__ __launch_bounds__(32)
void attn_kernel(const _Float16* __restrict__ Q, const _Float16* __restrict__ Kp,
                 const _Float16* __restrict__ Vt, _Float16* __restrict__ AO) {
  int qt = blockIdx.x & 127;           // T/16 = 128 query tiles
  int bh = blockIdx.x >> 7;
  int h  = bh & 15;
  int b  = bh >> 4;
  int q0 = qt * 16;
  int lane = threadIdx.x;
  int lg   = (lane & 16) >> 1;         // 0 or 8
  int nn   = lane & 15;

  __shared__ _Float16 Pl[16 * 64];     // P tile relayout buffer (2 KB)

  const _Float16* Qb = Q  + (size_t)b * T_SEQ * C_DIM + (size_t)h * D_HEAD;
  const _Float16* Kb = Kp + (size_t)b * T_SEQ * C_DIM + (size_t)h * D_HEAD;
  const _Float16* Vb = Vt + (size_t)(b * H_NUM + h) * D_HEAD * T_SEQ;

  // Q fragments for d = 0..31 and 32..63 (row stride C)
  h16 aq0 = load_a_h16(Qb, C_DIM, q0, 0);
  h16 aq1 = load_a_h16(Qb, C_DIM, q0, 32);

  f8 o[4] = {};                        // 4 d-tiles of the output
  float mrow[8], lrow[8];
#pragma unroll
  for (int i = 0; i < 8; ++i) { mrow[i] = -1e30f; lrow[i] = 0.0f; }

  for (int kc = 0; kc < T_SEQ; kc += 64) {
    // ---- S = Q*K^T for 16 queries x 64 keys (4 tiles x 2 k-steps) ----
    f8 s[4] = {};
#pragma unroll
    for (int j = 0; j < 4; ++j) {
      h16 bk = load_b_h16(Kb, C_DIM, kc + 16 * j, 0);
      s[j] = wmma_f16(aq0, bk, s[j]);
      bk = load_b_h16(Kb, C_DIM, kc + 16 * j, 32);
      s[j] = wmma_f16(aq1, bk, s[j]);
    }

    // ---- prefetch next chunk of K rows / V columns (global_prefetch_b8) ----
    if (kc + 64 < T_SEQ) {
      const _Float16* pk = Kb + (size_t)(kc + 64 + lane) * C_DIM;
      __builtin_prefetch(pk, 0, 0);
      __builtin_prefetch(pk + (size_t)32 * C_DIM, 0, 0);
      const _Float16* pv = Vb + (size_t)lane * T_SEQ + kc + 64;
      __builtin_prefetch(pv, 0, 0);
      __builtin_prefetch(pv + (size_t)32 * T_SEQ, 0, 0);
    }

    // ---- online softmax in exp2 domain; row m = i + lg lives on one
    //      vgpr across the 16 lanes of a half-group -> xor 1/2/4/8 ----
#pragma unroll
    for (int i = 0; i < 8; ++i) {
      float a0 = s[0][i] * ATTN_SCALE_LOG2;
      float a1 = s[1][i] * ATTN_SCALE_LOG2;
      float a2 = s[2][i] * ATTN_SCALE_LOG2;
      float a3 = s[3][i] * ATTN_SCALE_LOG2;
      float mx = fmaxf(fmaxf(a0, a1), fmaxf(a2, a3));
      mx = fmaxf(mx, __shfl_xor(mx, 1));
      mx = fmaxf(mx, __shfl_xor(mx, 2));
      mx = fmaxf(mx, __shfl_xor(mx, 4));
      mx = fmaxf(mx, __shfl_xor(mx, 8));
      float mnew  = fmaxf(mrow[i], mx);
      float alpha = exp2f(mrow[i] - mnew);
      mrow[i] = mnew;
      float p0 = exp2f(a0 - mnew);
      float p1 = exp2f(a1 - mnew);
      float p2 = exp2f(a2 - mnew);
      float p3 = exp2f(a3 - mnew);
      float rs = (p0 + p1) + (p2 + p3);
      rs += __shfl_xor(rs, 1);
      rs += __shfl_xor(rs, 2);
      rs += __shfl_xor(rs, 4);
      rs += __shfl_xor(rs, 8);
      lrow[i] = lrow[i] * alpha + rs;
      o[0][i] *= alpha; o[1][i] *= alpha; o[2][i] *= alpha; o[3][i] *= alpha;
      // stash P (C/D-layout element (m,n)) into LDS row-major 16x64
      int m = i + lg;
      Pl[m * 64 + nn]      = (_Float16)p0;
      Pl[m * 64 + nn + 16] = (_Float16)p1;
      Pl[m * 64 + nn + 32] = (_Float16)p2;
      Pl[m * 64 + nn + 48] = (_Float16)p3;
    }
    __syncthreads();

    // ---- reload P as two A-fragments (16x32 each) from LDS ----
    auto lds_a = [&](int k0) -> h16 {
      int r  = lane & 15;
      int kb = k0 + lg;                       // +0/8
      h8 lo = *(const h8*)&Pl[r * 64 + kb];
      h8 hi = *(const h8*)&Pl[r * 64 + kb + 16];
      h16 f;
#pragma unroll
      for (int i = 0; i < 8; ++i) { f[i] = lo[i]; f[i + 8] = hi[i]; }
      return f;
    };
    h16 pa0 = lds_a(0);
    h16 pa1 = lds_a(32);
    __syncthreads();

    // ---- O += P * V  (V transposed: row d, contiguous t) ----
#pragma unroll
    for (int j = 0; j < 4; ++j) {
      h16 bv = load_b_h16(Vb, T_SEQ, j * 16, kc);
      o[j] = wmma_f16(pa0, bv, o[j]);
      bv = load_b_h16(Vb, T_SEQ, j * 16, kc + 32);
      o[j] = wmma_f16(pa1, bv, o[j]);
    }
  }

  // ---- normalize (reciprocal once per row) and write (B,T,C) f16 ----
  float rinv[8];
#pragma unroll
  for (int i = 0; i < 8; ++i) rinv[i] = 1.0f / lrow[i];
#pragma unroll
  for (int j = 0; j < 4; ++j) {
#pragma unroll
    for (int i = 0; i < 8; ++i) {
      int m = i + lg;
      int n = j * 16 + nn;
      AO[((size_t)b * T_SEQ + q0 + m) * C_DIM + (size_t)h * D_HEAD + n] =
          (_Float16)(o[j][i] * rinv[i]);
    }
  }
}

// ---------------------------------------------------------------------------
extern "C" void kernel_launch(void* const* d_in, const int* in_sizes, int n_in,
                              void* d_out, int out_size, void* d_ws, size_t ws_size,
                              hipStream_t stream) {
  const float* q  = (const float*)d_in[0];
  const float* k  = (const float*)d_in[1];
  const float* v  = (const float*)d_in[2];
  const float* Wq = (const float*)d_in[3];
  const float* bq = (const float*)d_in[4];
  const float* Wk = (const float*)d_in[5];
  const float* bk = (const float*)d_in[6];
  const float* Wv = (const float*)d_in[7];
  const float* bv = (const float*)d_in[8];
  const float* Wo = (const float*)d_in[9];
  const float* bo = (const float*)d_in[10];
  float* out = (float*)d_out;

  const size_t WEL = (size_t)C_DIM * C_DIM;    // 1M elements
  const size_t AEL = (size_t)M_ROWS * C_DIM;   // 8M elements
  _Float16* Wq_h = (_Float16*)d_ws;
  _Float16* Wk_h = Wq_h + WEL;
  _Float16* Wv_h = Wk_h + WEL;
  _Float16* Wo_h = Wv_h + WEL;
  _Float16* Qp   = Wo_h + WEL;
  _Float16* Kp   = Qp + AEL;
  _Float16* Vt   = Kp + AEL;
  _Float16* AO   = Vt + AEL;

  // 1) weights to f16
  cvt_f32_f16<<<4096, 256, 0, stream>>>(Wq, Wq_h, (int)WEL);
  cvt_f32_f16<<<4096, 256, 0, stream>>>(Wk, Wk_h, (int)WEL);
  cvt_f32_f16<<<4096, 256, 0, stream>>>(Wv, Wv_h, (int)WEL);
  cvt_f32_f16<<<4096, 256, 0, stream>>>(Wo, Wo_h, (int)WEL);

  // 2) projections (4096 waves = 256 row tiles x 16 col tiles)
  proj_gemm<0, float><<<1024, 128, 0, stream>>>(q, Wq_h, bq, Qp);
  proj_gemm<0, float><<<1024, 128, 0, stream>>>(k, Wk_h, bk, Kp);
  proj_gemm<1, float><<<1024, 128, 0, stream>>>(v, Wv_h, bv, Vt);

  // 3) attention: 4 * 16 * 128 = 8192 single-wave workgroups
  attn_kernel<<<8192, 32, 0, stream>>>(Qp, Kp, Vt, AO);

  // 4) output projection to f32
  proj_gemm<2, _Float16><<<1024, 128, 0, stream>>>(AO, Wo_h, bo, out);
}
